// DSTFormer_50096498540651
// MI455X (gfx1250) — matrix operand order repaired and checked by reading, not compiled, not run on hardware
//
#include <hip/hip_runtime.h>
#include <hip/hip_bf16.h>

#define B_     16
#define T_     12
#define N_     300
#define D_     64
#define C_     64
#define M_     900      // 3*N
#define MP_    928      // padded to 32
#define NSTEP_ 29       // key steps of 32
#define THREADS_ 512
#define WAVES_   16

typedef _Float16 v8h  __attribute__((ext_vector_type(8)));
typedef _Float16 v16h __attribute__((ext_vector_type(16)));
typedef float    v8f  __attribute__((ext_vector_type(8)));

__device__ __forceinline__ v8f wmma_f16(v16h a, v16h b, v8f c) {
    // D = A(16x32 f16) x B(32x16 f16) + C(16x16 f32)
    return __builtin_amdgcn_wmma_f32_16x16x32_f16(false, a, false, b,
                                                  (short)0, c, false, false);
}

// B-operand fragment: B[k,n], n=lane&15; K-octets at (lane>>4)*8 and +16.
// "base" points at B tile stored so that column n is a contiguous run at base+n*stride+k.
__device__ __forceinline__ v16h load_b16(const _Float16* base, int stride, int lane) {
    const int n  = lane & 15;
    const int kb = (lane >> 4) * 8;
    const _Float16* p = base + n * stride + kb;
    v8h lo = *(const v8h*)(p);
    v8h hi = *(const v8h*)(p + 16);
    return __builtin_shufflevector(lo, hi, 0,1,2,3,4,5,6,7,8,9,10,11,12,13,14,15);
}

// A-operand fragment from a row-major f16 LDS tile (rows = M, cols = K).
// lane holds M=lane&15; K-octets at k0+(lane>>4)*8 and +16.
__device__ __forceinline__ v16h load_a16_lds(const _Float16* base, int stride,
                                             int k0, int lane) {
    const int mloc = lane & 15;
    const int kb   = k0 + (lane >> 4) * 8;
    const _Float16* p = base + mloc * stride + kb;
    v8h lo = *(const v8h*)(p);
    v8h hi = *(const v8h*)(p + 16);
    return __builtin_shufflevector(lo, hi, 0,1,2,3,4,5,6,7,8,9,10,11,12,13,14,15);
}

// A-operand fragment for stx row m0+ (lane&15), chunk d0 in {0,32}, straight from global
// fp32 x with replicate-padded sliding window; f32->f16 convert. Rows >= 900 are zero.
__device__ __forceinline__ v16h load_x_frag(const float* __restrict__ x, int bb, int t,
                                            int m0, int d0, int lane) {
    const int m  = m0 + (lane & 15);
    const int kb = d0 + (lane >> 4) * 8;
    v16h a;
    if (m < M_) {
        const int w  = m / N_;
        const int nn = m - w * N_;
        int ts = t - 1 + w;
        ts = ts < 0 ? 0 : (ts > T_ - 1 ? T_ - 1 : ts);
        const float* row = x + ((((size_t)bb * T_) + ts) * N_ + nn) * D_;
        const float4 f0 = *(const float4*)(row + kb);
        const float4 f1 = *(const float4*)(row + kb + 4);
        const float4 f2 = *(const float4*)(row + kb + 16);
        const float4 f3 = *(const float4*)(row + kb + 20);
        a[0]=(_Float16)f0.x; a[1]=(_Float16)f0.y; a[2]=(_Float16)f0.z; a[3]=(_Float16)f0.w;
        a[4]=(_Float16)f1.x; a[5]=(_Float16)f1.y; a[6]=(_Float16)f1.z; a[7]=(_Float16)f1.w;
        a[8]=(_Float16)f2.x; a[9]=(_Float16)f2.y; a[10]=(_Float16)f2.z; a[11]=(_Float16)f2.w;
        a[12]=(_Float16)f3.x; a[13]=(_Float16)f3.y; a[14]=(_Float16)f3.z; a[15]=(_Float16)f3.w;
    } else {
#pragma unroll
        for (int e = 0; e < 16; ++e) a[e] = (_Float16)0.f;
    }
    return a;
}

// Store a 16x16 f32 accumulator tile as f16 into a row-major LDS tile (base pre-offset by col).
__device__ __forceinline__ void store_tile(_Float16* base, int stride, v8f acc, int lane) {
    const int c  = lane & 15;
    const int ms = (lane >> 4) * 8;
#pragma unroll
    for (int r = 0; r < 8; ++r) base[(ms + r) * stride + c] = (_Float16)acc[r];
}

// --------- prologue: precompute bit-packed attention mask (same for every (b,t)) ---------
// wsmask[m*NSTEP_+j] bit b == 1  <=>  key l=j*32+b is masked for query m.
__global__ void dstformer_mask(const unsigned char* __restrict__ geo,
                               unsigned int* __restrict__ wsmask) {
    const int idx = blockIdx.x * blockDim.x + threadIdx.x;
    if (idx >= MP_ * NSTEP_) return;
    const int m = idx / NSTEP_;
    const int j = idx - m * NSTEP_;
    if (m >= M_) { wsmask[idx] = 0xFFFFFFFFu; return; }
    const int wi = m / N_, ni = m - (m / N_) * N_;
    unsigned bits = 0u;
#pragma unroll 4
    for (int b = 0; b < 32; ++b) {
        const int l = j * 32 + b;
        bool msk;
        if (l >= M_) {
            msk = true;
        } else {
            const int wj = l / N_, nj = l - wj * N_;
            if (wi == wj)                       msk = geo[ni * N_ + nj] != 0;
            else if (wi == wj + 1 || wj == wi + 1) msk = (ni != nj);
            else                                msk = true;
        }
        if (msk) bits |= (1u << b);
    }
    wsmask[idx] = bits;
}

// --------- main fused kernel: one workgroup per (b,t) ---------
__global__ __launch_bounds__(THREADS_)
void dstformer_attn(const float* __restrict__ x,
                    const float* __restrict__ Wq,
                    const float* __restrict__ Wk,
                    const float* __restrict__ Wv,
                    const unsigned int* __restrict__ wsmask,
                    float* __restrict__ out) {
    extern __shared__ _Float16 smem[];
    _Float16* Wqs = smem;                 // 64*64
    _Float16* Wks = Wqs + 64 * 64;        // 64*64
    _Float16* Wvs = Wks + 64 * 64;        // 64*64
    _Float16* Kl  = Wvs + 64 * 64;        // [MP_][64]   row-major K
    _Float16* Vt  = Kl  + MP_ * 64;       // [64][MP_]   V transposed
    _Float16* scr = Vt  + 64 * MP_;       // per-wave 16x64 scratch

    const int tid  = threadIdx.x;
    const int lane = tid & 31;
    const int wave = tid >> 5;
    const int bt   = blockIdx.x;
    const int bb   = bt / T_;
    const int t    = bt - bb * T_;
    const int lo   = lane & 15;
    const int hi   = lane >> 4;
    const float NEG_INF = -__builtin_inff();

    // stage weights (f32 -> f16) in LDS; B-operand wants W row-major ([c][d]).
    for (int i = tid; i < 64 * 64; i += THREADS_) {
        Wqs[i] = (_Float16)Wq[i];
        Wks[i] = (_Float16)Wk[i];
        Wvs[i] = (_Float16)Wv[i];
    }
    __syncthreads();

    // ---------- Phase A: K (row-major) and V^T into LDS ----------
    for (int tile = wave; tile < MP_ / 16; tile += WAVES_) {
        const int m0 = tile * 16;
        v16h a0 = load_x_frag(x, bb, t, m0, 0,  lane);
        v16h a1 = load_x_frag(x, bb, t, m0, 32, lane);
#pragma unroll
        for (int cc = 0; cc < 4; ++cc) {
            v8f kacc = {};
            kacc = wmma_f16(a0, load_b16(Wks + cc * 16 * 64,      64, lane), kacc);
            kacc = wmma_f16(a1, load_b16(Wks + cc * 16 * 64 + 32, 64, lane), kacc);
            store_tile(Kl + m0 * 64 + cc * 16, 64, kacc, lane);

            v8f vacc = {};
            vacc = wmma_f16(a0, load_b16(Wvs + cc * 16 * 64,      64, lane), vacc);
            vacc = wmma_f16(a1, load_b16(Wvs + cc * 16 * 64 + 32, 64, lane), vacc);
#pragma unroll
            for (int r = 0; r < 8; ++r)
                Vt[(cc * 16 + lo) * MP_ + m0 + hi * 8 + r] = (_Float16)vacc[r];
        }
    }
    __syncthreads();

    // ---------- Phase B: per-wave flash attention over 16-query tiles ----------
    _Float16* myscr = scr + wave * 1024;   // 16x64 halves
    for (int tile = wave; tile < 57; tile += WAVES_) {   // rows 0..911 (>=900 discarded)
        const int m0 = tile * 16;

        // Q tile via WMMA, bounce through LDS to re-fragment into A layout
        {
            v16h a0 = load_x_frag(x, bb, t, m0, 0,  lane);
            v16h a1 = load_x_frag(x, bb, t, m0, 32, lane);
#pragma unroll
            for (int cc = 0; cc < 4; ++cc) {
                v8f q = {};
                q = wmma_f16(a0, load_b16(Wqs + cc * 16 * 64,      64, lane), q);
                q = wmma_f16(a1, load_b16(Wqs + cc * 16 * 64 + 32, 64, lane), q);
                store_tile(myscr + cc * 16, 64, q, lane);
            }
        }
        v16h qa0 = load_a16_lds(myscr, 64, 0,  lane);
        v16h qa1 = load_a16_lds(myscr, 64, 32, lane);

        v8f O0 = {}, O1 = {}, O2 = {}, O3 = {};
        float mi[8], li[8];
#pragma unroll
        for (int r = 0; r < 8; ++r) { mi[r] = NEG_INF; li[r] = 0.f; }

        for (int j = 0; j < NSTEP_; ++j) {
            const int l0 = j * 32;
            v8f s0 = {}, s1 = {};
            s0 = wmma_f16(qa0, load_b16(Kl + l0 * 64,             64, lane), s0);
            s0 = wmma_f16(qa1, load_b16(Kl + l0 * 64 + 32,        64, lane), s0);
            s1 = wmma_f16(qa0, load_b16(Kl + (l0 + 16) * 64,      64, lane), s1);
            s1 = wmma_f16(qa1, load_b16(Kl + (l0 + 16) * 64 + 32, 64, lane), s1);

#pragma unroll
            for (int r = 0; r < 8; ++r) {
                const int mrow = m0 + hi * 8 + r;
                const unsigned mb = wsmask[mrow * NSTEP_ + j];
                float av = ((mb >> lo)        & 1u) ? NEG_INF : s0[r];
                float bv = ((mb >> (16 + lo)) & 1u) ? NEG_INF : s1[r];
                float rm = fmaxf(av, bv);
                rm = fmaxf(rm, __shfl_xor(rm, 1, 32));
                rm = fmaxf(rm, __shfl_xor(rm, 2, 32));
                rm = fmaxf(rm, __shfl_xor(rm, 4, 32));
                rm = fmaxf(rm, __shfl_xor(rm, 8, 32));
                const float nm = fmaxf(mi[r], rm);
                float scale, p0, p1;
                if (nm == NEG_INF) { scale = 1.f; p0 = 0.f; p1 = 0.f; }
                else {
                    scale = __expf(mi[r] - nm);
                    p0 = __expf(av - nm);
                    p1 = __expf(bv - nm);
                }
                mi[r] = nm;
                float rs = p0 + p1;
                rs += __shfl_xor(rs, 1, 32);
                rs += __shfl_xor(rs, 2, 32);
                rs += __shfl_xor(rs, 4, 32);
                rs += __shfl_xor(rs, 8, 32);
                li[r] = li[r] * scale + rs;
                O0[r] *= scale; O1[r] *= scale; O2[r] *= scale; O3[r] *= scale;
                myscr[(hi * 8 + r) * 32 + lo]      = (_Float16)p0;
                myscr[(hi * 8 + r) * 32 + 16 + lo] = (_Float16)p1;
            }
            v16h pa = load_a16_lds(myscr, 32, 0, lane);
            O0 = wmma_f16(pa, load_b16(Vt + 0 * 16 * MP_ + l0, MP_, lane), O0);
            O1 = wmma_f16(pa, load_b16(Vt + 1 * 16 * MP_ + l0, MP_, lane), O1);
            O2 = wmma_f16(pa, load_b16(Vt + 2 * 16 * MP_ + l0, MP_, lane), O2);
            O3 = wmma_f16(pa, load_b16(Vt + 3 * 16 * MP_ + l0, MP_, lane), O3);
        }

        // normalize + scattered store implementing swapaxes(2,3).reshape(...)[:, :, 1]
        float* outbt = out + (size_t)(bb * T_ + t) * N_ * C_;
#pragma unroll
        for (int r = 0; r < 8; ++r) {
            const int m = m0 + hi * 8 + r;
            if (m < M_) {
                const float inv = li[r] > 0.f ? 1.f / li[r] : 0.f;
#pragma unroll
                for (int cc = 0; cc < 4; ++cc) {
                    const float yv =
                        (cc == 0 ? O0[r] : cc == 1 ? O1[r] : cc == 2 ? O2[r] : O3[r]) * inv;
                    const int c    = cc * 16 + lo;
                    const int flat = c * M_ + m;     // y^T flat index
                    const int i    = flat >> 6;
                    if (i >= N_ && i < 2 * N_)
                        outbt[(i - N_) * 64 + (flat & 63)] = yv;
                }
            }
        }
    }
}

extern "C" void kernel_launch(void* const* d_in, const int* in_sizes, int n_in,
                              void* d_out, int out_size, void* d_ws, size_t ws_size,
                              hipStream_t stream) {
    const float*         x   = (const float*)d_in[0];
    const unsigned char* geo = (const unsigned char*)d_in[1];  // jnp bool -> 1 byte
    const float*         Wq  = (const float*)d_in[2];
    const float*         Wk  = (const float*)d_in[3];
    const float*         Wv  = (const float*)d_in[4];
    float*               out = (float*)d_out;
    unsigned int*     wsmask = (unsigned int*)d_ws;            // MP_*NSTEP_*4 = 107,648 B

    const int nmask = MP_ * NSTEP_;
    dstformer_mask<<<(nmask + 255) / 256, 256, 0, stream>>>(geo, wsmask);

    const size_t smem_bytes =
        (size_t)(3 * 64 * 64 + MP_ * 64 + 64 * MP_ + WAVES_ * 1024) * sizeof(_Float16);
    (void)hipFuncSetAttribute((const void*)dstformer_attn,
                              hipFuncAttributeMaxDynamicSharedMemorySize,
                              (int)smem_bytes);   // 294,912 B < 320 KB WGP budget

    dstformer_attn<<<B_ * T_, THREADS_, smem_bytes, stream>>>(x, Wq, Wk, Wv, wsmask, out);
}
